// GCN_DAE_32255204393056
// MI455X (gfx1250) — matrix operand order repaired and checked by reading, not compiled, not run on hardware
//
#include <hip/hip_runtime.h>
#include <hip/hip_bf16.h>

typedef __attribute__((ext_vector_type(16))) _Float16 v16h;
typedef __attribute__((ext_vector_type(8)))  _Float16 v8h;
typedef __attribute__((ext_vector_type(8)))  float    v8f;

#define NROWS 8192
#define KSEL  31   // top_k(adj, k+1) with k=30

// ---------------------------------------------------------------------------
// f32 -> f16 elementwise convert (grid-stride)
// ---------------------------------------------------------------------------
__global__ __launch_bounds__(256) void f32_to_f16(const float* __restrict__ in,
                                                  _Float16* __restrict__ out,
                                                  size_t n) {
  size_t i = (size_t)blockIdx.x * blockDim.x + threadIdx.x;
  size_t stride = (size_t)gridDim.x * blockDim.x;
  for (; i < n; i += stride) out[i] = (_Float16)in[i];
}

// ---------------------------------------------------------------------------
// Generic WMMA GEMM: C[M,N] = A[M,K] @ Bt[N,K]^T  (f16 in, f32 accumulate)
// Each wave computes a 16 x (NT*16) strip: one A fragment is reused for NT
// WMMAs (register blocking along N). The K-loop is unrolled 2x with ping-pong
// fragment buffers: loads for step k+64 write directly into the buffer that
// step k's WMMAs just consumed (no register copies), so each s_wait_loadcnt
// drains only one buffer while the other's loads remain in flight.
// REQUIRES: K a multiple of 64 (all GEMMs here use K in {256, 512, 8192}).
// ---------------------------------------------------------------------------
template<int NT, bool RELU, bool OUT_F16, bool OUT_TR, bool HAS_BIAS>
__global__ __launch_bounds__(128) void wmma_gemm_abt(
    const _Float16* __restrict__ A, const _Float16* __restrict__ Bt,
    const float* __restrict__ bias, float* __restrict__ Cf,
    _Float16* __restrict__ Ch, int M, int N, int K, int ldc) {
  const int wave = threadIdx.x >> 5;
  const int lane = threadIdx.x & 31;
  const int half = lane >> 4;   // 0: lanes 0-15, 1: lanes 16-31
  const int r    = lane & 15;
  const int m0 = (blockIdx.y * 4 + wave) * 16;
  const int n0 = blockIdx.x * (NT * 16);
  if (m0 >= M || n0 >= N) return;

  // A fragment (16-bit A 16x32): lane holds row m0+r.
  //   halves[0..15] = K = kk + half*16 + {0..15}   (contiguous)
  // B fragment (B = Bt^T, 32x16): lane holds column n+r of C, i.e. row n+r
  //   of Bt; halves[0..15] = K = kk + half*16 + {0..15}   (contiguous!)
  const _Float16* arow = A + (size_t)(m0 + r) * K;
  const _Float16* brow[NT];
#pragma unroll
  for (int t = 0; t < NT; ++t)
    brow[t] = Bt + (size_t)(n0 + t * 16 + r) * K;
  const int kA = half * 16;
  const int kB = half * 16;

  v8f acc[NT];
#pragma unroll
  for (int t = 0; t < NT; ++t) acc[t] = (v8f)(0.0f);

  auto loadFrag = [&](int kk, v16h& a, v16h (&b)[NT]) {
    a = *(const v16h*)(arow + kk + kA);
#pragma unroll
    for (int t = 0; t < NT; ++t)
      b[t] = *(const v16h*)(brow[t] + kk + kB);
  };
  auto mmaFrag = [&](const v16h& a, const v16h (&b)[NT]) {
#pragma unroll
    for (int t = 0; t < NT; ++t)
      acc[t] = __builtin_amdgcn_wmma_f32_16x16x32_f16(
          false, a, false, b[t], (short)0, acc[t], false, false);
  };

  v16h a0, a1;
  v16h b0[NT], b1[NT];
  loadFrag(0, a0, b0);
  loadFrag(32, a1, b1);

  const int PF = 1024;  // prefetch distance in halves (2KB)
  int kk = 0;
  for (; kk + 64 < K; kk += 64) {
    mmaFrag(a0, b0);            // waits only on buffer 0's loads
    loadFrag(kk + 64, a0, b0);  // refill buffer 0 in place
    if (kk + PF < K) __builtin_prefetch(arow + kk + PF, 0, 1);
    mmaFrag(a1, b1);            // waits only on buffer 1's loads
    loadFrag(kk + 96, a1, b1);  // refill buffer 1 in place
  }
  mmaFrag(a0, b0);              // tail: last two K-steps
  mmaFrag(a1, b1);

#pragma unroll
  for (int t = 0; t < NT; ++t) {
    const int n = n0 + t * 16 + r;
    float bv = 0.0f;
    if (HAS_BIAS) bv = bias[n];
    if (OUT_TR) {
      // C element (m0+v+8*half, n) -> Ct[n][m0+v+8*half]; v contiguous.
      v8h pack;
#pragma unroll
      for (int v = 0; v < 8; ++v) {
        float val = acc[t][v] + bv;
        if (RELU) val = fmaxf(val, 0.0f);
        pack[v] = (_Float16)val;
      }
      *(v8h*)(Ch + (size_t)n * ldc + m0 + 8 * half) = pack;
    } else {
#pragma unroll
      for (int v = 0; v < 8; ++v) {
        float val = acc[t][v] + bv;
        if (RELU) val = fmaxf(val, 0.0f);
        size_t off = (size_t)(m0 + v + 8 * half) * ldc + n;
        if (OUT_F16) Ch[off] = (_Float16)val;
        else         Cf[off] = val;
      }
    }
  }
}

// ---------------------------------------------------------------------------
// Row L2-normalize: emb = h / max(||h||, 1e-12), emitted as f16
// ---------------------------------------------------------------------------
__global__ __launch_bounds__(256) void row_normalize_f16(
    const float* __restrict__ H, _Float16* __restrict__ E, int K) {
  const int row = blockIdx.x;
  const int tid = threadIdx.x;
  const float* h = H + (size_t)row * K;
  float s = 0.0f;
  for (int j = tid; j < K; j += 256) { float v = h[j]; s += v * v; }
  __shared__ float red[256];
  red[tid] = s;
  __syncthreads();
  for (int off = 128; off > 0; off >>= 1) {
    if (tid < off) red[tid] += red[tid + off];
    __syncthreads();
  }
  const float inv = 1.0f / fmaxf(sqrtf(red[0]), 1e-12f);
  for (int j = tid; j < K; j += 256)
    E[(size_t)row * K + j] = (_Float16)(h[j] * inv);
}

// ---------------------------------------------------------------------------
// Per-row top-KSEL threshold + relu, in place. 64 threads/row.
// Each thread keeps a sorted local top-31 of its strided chunk; the global
// 31st-largest is extracted from the 64*31 candidates in LDS.
// ---------------------------------------------------------------------------
__global__ __launch_bounds__(64) void topk_threshold_relu(
    float* __restrict__ S, int N) {
  const int row = blockIdx.x;
  const int tid = threadIdx.x;
  float* srow = S + (size_t)row * N;

  float loc[KSEL];
#pragma unroll
  for (int i = 0; i < KSEL; ++i) loc[i] = -3.0e38f;
  for (int j = tid; j < N; j += 64) {
    float v = srow[j];
    if (v > loc[KSEL - 1]) {
      int p = KSEL - 1;
      while (p > 0 && loc[p - 1] < v) { loc[p] = loc[p - 1]; --p; }
      loc[p] = v;
    }
  }

  __shared__ float cand[64 * KSEL];
  __shared__ float sv[64];
  __shared__ int   si[64];
  __shared__ float sthresh;
#pragma unroll
  for (int i = 0; i < KSEL; ++i) cand[tid * KSEL + i] = loc[i];
  __syncthreads();

  for (int round = 0; round < KSEL; ++round) {
    float bv = -3.0e38f; int bidx = 0;
    for (int i = tid; i < 64 * KSEL; i += 64) {
      float c = cand[i];
      if (c > bv) { bv = c; bidx = i; }
    }
    sv[tid] = bv; si[tid] = bidx;
    __syncthreads();
    if (tid == 0) {
      float m = -3.0e38f; int mi = 0;
      for (int t = 0; t < 64; ++t)
        if (sv[t] > m) { m = sv[t]; mi = si[t]; }
      cand[mi] = -3.0e38f;
      if (round == KSEL - 1) sthresh = m;
    }
    __syncthreads();
  }
  const float t = sthresh;
  for (int j = tid; j < N; j += 64) {
    float v = srow[j];
    srow[j] = (v >= t) ? fmaxf(v, 0.0f) : 0.0f;  // relu(top_k_mask(S))
  }
}

// ---------------------------------------------------------------------------
// In-place symmetrize A = 0.5*(A + A^T), 64x64 tile pairs via LDS.
// ---------------------------------------------------------------------------
#define ST 64
__global__ __launch_bounds__(256) void symmetrize_inplace(
    float* __restrict__ Adj, int N) {
  const int bi = blockIdx.y, bj = blockIdx.x;
  if (bj < bi) return;
  __shared__ float T1[ST][ST + 1];
  __shared__ float T2[ST][ST + 1];
  const int tid = threadIdx.x;
  const size_t base1 = (size_t)bi * ST * N + (size_t)bj * ST;
  const size_t base2 = (size_t)bj * ST * N + (size_t)bi * ST;
  for (int l = tid; l < ST * ST; l += 256) {
    int i = l >> 6, j = l & 63;
    T1[i][j] = Adj[base1 + (size_t)i * N + j];
    T2[i][j] = Adj[base2 + (size_t)i * N + j];
  }
  __syncthreads();
  for (int l = tid; l < ST * ST; l += 256) {
    int i = l >> 6, j = l & 63;
    Adj[base1 + (size_t)i * N + j] = 0.5f * (T1[i][j] + T2[j][i]);
  }
  if (bi != bj) {
    for (int l = tid; l < ST * ST; l += 256) {
      int i = l >> 6, j = l & 63;
      Adj[base2 + (size_t)i * N + j] = 0.5f * (T2[i][j] + T1[j][i]);
    }
  }
}

// ---------------------------------------------------------------------------
// invs[i] = 1/(sqrt(rowsum) + 1e-10)
// ---------------------------------------------------------------------------
__global__ __launch_bounds__(256) void row_inv_sqrt_deg(
    const float* __restrict__ Adj, float* __restrict__ invs, int N) {
  const int row = blockIdx.x;
  const int tid = threadIdx.x;
  const float* a = Adj + (size_t)row * N;
  float s = 0.0f;
  for (int j = tid; j < N; j += 256) s += a[j];
  __shared__ float red[256];
  red[tid] = s;
  __syncthreads();
  for (int off = 128; off > 0; off >>= 1) {
    if (tid < off) red[tid] += red[tid + off];
    __syncthreads();
  }
  if (tid == 0) invs[row] = 1.0f / (sqrtf(red[0]) + 1e-10f);
}

// ---------------------------------------------------------------------------
// adj[i,j] *= invs[i]*invs[j] in place; also emit f16 copy. N == 8192.
// ---------------------------------------------------------------------------
__global__ __launch_bounds__(256) void scale_and_half(
    float* __restrict__ Adj, _Float16* __restrict__ AdjH,
    const float* __restrict__ invs) {
  const size_t idx = (size_t)blockIdx.x * 256 + threadIdx.x;
  const int i = (int)(idx >> 13);      // /8192
  const int j = (int)(idx & 8191);     // %8192
  const float v = Adj[idx] * invs[i] * invs[j];
  Adj[idx]  = v;
  AdjH[idx] = (_Float16)v;
}

// ---------------------------------------------------------------------------
extern "C" void kernel_launch(void* const* d_in, const int* in_sizes, int n_in,
                              void* d_out, int out_size, void* d_ws,
                              size_t ws_size, hipStream_t stream) {
  (void)in_sizes; (void)n_in; (void)out_size; (void)ws_size;
  const float* features = (const float*)d_in[0];  // [8192,512]
  const float* x        = (const float*)d_in[1];  // [8192,512]
  const float* gsl_W0   = (const float*)d_in[2];  // [512,512]
  const float* gsl_b0   = (const float*)d_in[3];
  const float* gsl_W1   = (const float*)d_in[4];  // [512,512]
  const float* gsl_b1   = (const float*)d_in[5];
  const float* gcn_W0   = (const float*)d_in[6];  // [256,512]
  const float* gcn_b0   = (const float*)d_in[7];
  const float* gcn_W1   = (const float*)d_in[8];  // [16,256]
  const float* gcn_b1   = (const float*)d_in[9];

  float* x2_out  = (float*)d_out;                               // [8192,16]
  float* adj_out = (float*)d_out + (size_t)NROWS * 16;          // [8192,8192]

  // ---- workspace carve-out (256B aligned) ----
  char* ws = (char*)d_ws;
  size_t off = 0;
  auto alloc = [&](size_t bytes) {
    char* p = ws + off;
    off += (bytes + 255) & ~(size_t)255;
    return p;
  };
  _Float16* featH = (_Float16*)alloc((size_t)NROWS * 512 * 2);
  _Float16* xH    = (_Float16*)alloc((size_t)NROWS * 512 * 2);
  _Float16* w0H   = (_Float16*)alloc((size_t)512 * 512 * 2);
  _Float16* w1H   = (_Float16*)alloc((size_t)512 * 512 * 2);
  _Float16* gw0H  = (_Float16*)alloc((size_t)256 * 512 * 2);
  _Float16* gw1H  = (_Float16*)alloc((size_t)16 * 256 * 2);
  _Float16* H0H   = (_Float16*)alloc((size_t)NROWS * 512 * 2);
  float*    H1f   = (float*)   alloc((size_t)NROWS * 512 * 4);
  _Float16* embH  = (_Float16*)alloc((size_t)NROWS * 512 * 2);
  float*    invs  = (float*)   alloc((size_t)NROWS * 4);
  _Float16* Yt    = (_Float16*)alloc((size_t)256 * NROWS * 2);  // Y^T
  _Float16* X1H   = (_Float16*)alloc((size_t)NROWS * 256 * 2);
  _Float16* Y2t   = (_Float16*)alloc((size_t)16 * NROWS * 2);   // Y2^T
  _Float16* adjH  = (_Float16*)alloc((size_t)NROWS * NROWS * 2);

  // ---- 1. convert inputs to f16 ----
  f32_to_f16<<<2048, 256, 0, stream>>>(features, featH, (size_t)NROWS * 512);
  f32_to_f16<<<2048, 256, 0, stream>>>(x, xH, (size_t)NROWS * 512);
  f32_to_f16<<<512, 256, 0, stream>>>(gsl_W0, w0H, (size_t)512 * 512);
  f32_to_f16<<<512, 256, 0, stream>>>(gsl_W1, w1H, (size_t)512 * 512);
  f32_to_f16<<<256, 256, 0, stream>>>(gcn_W0, gw0H, (size_t)256 * 512);
  f32_to_f16<<<16, 256, 0, stream>>>(gcn_W1, gw1H, (size_t)16 * 256);

  const dim3 blk(128);
  const int my = NROWS / 64;  // 128 blocks along M (4 waves * 16 rows each)

  // ---- 2. H0 = relu(features @ W0^T + b0), f16 ----
  wmma_gemm_abt<4, true, true, false, true>
      <<<dim3(512 / 64, my), blk, 0, stream>>>(
      featH, w0H, gsl_b0, nullptr, H0H, NROWS, 512, 512, 512);
  // ---- 3. H1 = H0 @ W1^T + b1, f32 ----
  wmma_gemm_abt<4, false, false, false, true>
      <<<dim3(512 / 64, my), blk, 0, stream>>>(
      H0H, w1H, gsl_b1, H1f, nullptr, NROWS, 512, 512, 512);
  // ---- 4. emb = normalize(H1), f16 ----
  row_normalize_f16<<<NROWS, 256, 0, stream>>>(H1f, embH, 512);
  // ---- 5. S = emb @ emb^T, straight into d_out adj region ----
  wmma_gemm_abt<4, false, false, false, false>
      <<<dim3(NROWS / 64, my), blk, 0, stream>>>(
      embH, embH, nullptr, adj_out, nullptr, NROWS, NROWS, 512, NROWS);
  // ---- 6. per-row top-31 mask + relu (in place) ----
  topk_threshold_relu<<<NROWS, 64, 0, stream>>>(adj_out, NROWS);
  // ---- 7. A = 0.5*(A + A^T) (in place) ----
  symmetrize_inplace<<<dim3(NROWS / ST, NROWS / ST), 256, 0, stream>>>(adj_out, NROWS);
  // ---- 8. degree scaling factors ----
  row_inv_sqrt_deg<<<NROWS, 256, 0, stream>>>(adj_out, invs, NROWS);
  // ---- 9. adj = D^-1/2 A D^-1/2 (in place) + f16 copy ----
  scale_and_half<<<(unsigned)((size_t)NROWS * NROWS / 256), 256, 0, stream>>>(
      adj_out, adjH, invs);
  // ---- 10. Y^T = (x @ gcn_W0^T + b0)^T, f16 [256, 8192] ----
  wmma_gemm_abt<4, false, true, true, true>
      <<<dim3(256 / 64, my), blk, 0, stream>>>(
      xH, gw0H, gcn_b0, nullptr, Yt, NROWS, 256, 512, NROWS);
  // ---- 11. X1 = relu(adj @ Y), f16 [8192, 256] ----
  wmma_gemm_abt<4, true, true, false, false>
      <<<dim3(256 / 64, my), blk, 0, stream>>>(
      adjH, Yt, nullptr, nullptr, X1H, NROWS, 256, NROWS, 256);
  // ---- 12. Y2^T = (X1 @ gcn_W1^T + b1)^T, f16 [16, 8192] ----
  wmma_gemm_abt<1, false, true, true, true>
      <<<dim3(1, my), blk, 0, stream>>>(
      X1H, gw1H, gcn_b1, nullptr, Y2t, NROWS, 16, 256, NROWS);
  // ---- 13. x2 = adj @ Y2, f32 -> d_out ----
  wmma_gemm_abt<1, false, false, false, false>
      <<<dim3(1, my), blk, 0, stream>>>(
      adjH, Y2t, nullptr, x2_out, nullptr, NROWS, 16, NROWS, 16);
}
